// LocalGlobalRegistrationWithDuplicateRemoval_2173253452329
// MI455X (gfx1250) — matrix operand order repaired and checked by reading, not compiled, not run on hardware
//
#include <hip/hip_runtime.h>
#include <hip/hip_bf16.h>
#include <stdint.h>

// ---------------- problem constants (match reference) ----------------
#define NPTS 4096
#define MPTS 4096
#define NM   16777216          // NPTS*MPTS
#define BB   256
#define KK   64
#define GG   4096              // MAX_GLOBAL
#define RAD2 4.0f              // RADIUS^2
#define EPSW 1e-5f
#define NSORT 65536
#define HBINS 1024

typedef __attribute__((ext_vector_type(16))) _Float16 v16h;
typedef __attribute__((ext_vector_type(8)))  float    v8f;

// ---------------- 3x3 SVD (Kabsch) helpers ----------------
__device__ inline float det3(const float m[3][3]) {
  return m[0][0]*(m[1][1]*m[2][2]-m[1][2]*m[2][1])
       - m[0][1]*(m[1][0]*m[2][2]-m[1][2]*m[2][0])
       + m[0][2]*(m[1][0]*m[2][1]-m[1][1]*m[2][0]);
}

// R = V * diag(1,1,d) * U^T  where H = U S V^T (reference Kabsch)
__device__ void kabsch_from_H(const float H[3][3], float R[3][3]) {
  // A = H^T H, symmetric PSD
  float A[3][3];
  for (int i=0;i<3;i++) for (int j=0;j<3;j++){
    float s=0.f; for (int k=0;k<3;k++) s += H[k][i]*H[k][j];
    A[i][j]=s;
  }
  float V[3][3] = {{1,0,0},{0,1,0},{0,0,1}};
  for (int sweep=0; sweep<14; sweep++) {
    for (int p=0;p<2;p++) for (int q=p+1;q<3;q++) {
      float apq = A[p][q];
      if (fabsf(apq) < 1e-14f) continue;
      float theta = (A[q][q]-A[p][p]) / (2.f*apq);
      float tJ = copysignf(1.f, theta) / (fabsf(theta) + sqrtf(1.f + theta*theta));
      float c = 1.f / sqrtf(1.f + tJ*tJ);
      float s = tJ * c;
      float App=A[p][p], Aqq=A[q][q], Apq=A[p][q];
      A[p][p] = c*c*App - 2.f*s*c*Apq + s*s*Aqq;
      A[q][q] = s*s*App + 2.f*s*c*Apq + c*c*Aqq;
      A[p][q] = A[q][p] = 0.f;
      int r = 3 - p - q;
      float Arp = A[r][p], Arq = A[r][q];
      A[r][p] = A[p][r] = c*Arp - s*Arq;
      A[r][q] = A[q][r] = s*Arp + c*Arq;
      for (int kI=0;kI<3;kI++){
        float vp=V[kI][p], vq=V[kI][q];
        V[kI][p] = c*vp - s*vq;
        V[kI][q] = s*vp + c*vq;
      }
    }
  }
  // sort eigenpairs descending
  float lam[3] = {A[0][0], A[1][1], A[2][2]};
  int ord[3] = {0,1,2};
  for (int i=0;i<2;i++) for (int j=i+1;j<3;j++)
    if (lam[ord[j]] > lam[ord[i]]) { int t=ord[i]; ord[i]=ord[j]; ord[j]=t; }
  float Vs[3][3], U[3][3];
  for (int c=0;c<3;c++){
    int o = ord[c];
    float sig = sqrtf(fmaxf(lam[o], 0.f));
    float inv = 1.f / fmaxf(sig, 1e-20f);
    for (int r=0;r<3;r++) Vs[r][c] = V[r][o];
    for (int r=0;r<3;r++){
      float u = 0.f;
      for (int k=0;k<3;k++) u += H[r][k]*V[k][o];
      U[r][c] = u * inv;
    }
  }
  float d = (det3(Vs)*det3(U) < 0.f) ? -1.f : 1.f;
  for (int i=0;i<3;i++) for (int j=0;j<3;j++)
    R[i][j] = Vs[i][0]*U[j][0] + Vs[i][1]*U[j][1] + d*Vs[i][2]*U[j][2];
}

// S: [0]=sum w; [1..3]=sum w*s; [4..6]=sum w*t; [7..15]=sum w*s_i*t_j (row major)
__device__ void procrustes_from_sums(const float* S, float R[3][3], float t[3]) {
  float k = 1.f / (S[0] + EPSW);
  float cs[3], ct[3];
  for (int i=0;i<3;i++){ cs[i]=k*S[1+i]; ct[i]=k*S[4+i]; }
  float c2 = 2.f - k*S[0];
  float H[3][3];
  for (int i=0;i<3;i++) for (int j=0;j<3;j++)
    H[i][j] = k*S[7+3*i+j] - cs[i]*ct[j]*c2;
  kabsch_from_H(H, R);
  for (int i=0;i<3;i++){
    float rc = 0.f;
    for (int j=0;j<3;j++) rc += R[i][j]*cs[j];
    t[i] = ct[i] - rc;
  }
}

__device__ inline float warp_sum(float v) {
  for (int o=16;o>0;o>>=1) v += __shfl_down(v, o, 32);
  return v;
}

// ---------------- kernel 1: scatter masked scores into pair table ----------------
__global__ void scatter_pairs(const int* __restrict__ sidx, const int* __restrict__ tidx,
                              const float* __restrict__ score, const unsigned char* __restrict__ corr,
                              float* __restrict__ tabS, float* __restrict__ tabC) {
  int tid = blockIdx.x*blockDim.x + threadIdx.x;   // 0 .. BB*KK*KK-1
  int b = tid >> 12, rem = tid & 4095, i = rem >> 6, j = rem & 63;
  if (corr[tid]) {
    int h = sidx[b*KK+i] * MPTS + tidx[b*KK+j];    // < 2^24, fits int
    atomicAdd(&tabS[h], score[tid]);
    atomicAdd(&tabC[h], 1.0f);
  }
}

// ---------------- kernel 2: histogram of avg scores ----------------
__global__ void hist_kernel(const float* __restrict__ tabS, const float* __restrict__ tabC,
                            unsigned* __restrict__ hist) {
  __shared__ unsigned lh[HBINS];
  for (int i=threadIdx.x;i<HBINS;i+=blockDim.x) lh[i]=0u;
  __syncthreads();
  long long stride = (long long)gridDim.x*blockDim.x;
  for (long long i = (long long)blockIdx.x*blockDim.x + threadIdx.x; i < NM; i += stride) {
    float tc = tabC[i];
    if (tc > 0.f) {
      float v = tabS[i] / fmaxf(tc, 1.f);
      int bin = min(HBINS-1, (int)(v * (float)HBINS));
      atomicAdd(&lh[bin], 1u);
    }
  }
  __syncthreads();
  for (int i=threadIdx.x;i<HBINS;i+=blockDim.x) if (lh[i]) atomicAdd(&hist[i], lh[i]);
}

// ---------------- kernel 3: pick threshold bin (largest bin with suffix >= GG) ----------------
__global__ void find_threshold(const unsigned* __restrict__ hist, unsigned* __restrict__ thrBin) {
  unsigned cum = 0; int t = 0;
  for (int b = HBINS-1; b >= 0; b--) { cum += hist[b]; if (cum >= GG) { t = b; break; } }
  *thrBin = (unsigned)t;
}

// ---------------- kernel 4: compact candidates >= threshold into sortable keys ----------------
__global__ void compact_kernel(const float* __restrict__ tabS, const float* __restrict__ tabC,
                               const unsigned* __restrict__ thrBin, unsigned* __restrict__ candCnt,
                               unsigned long long* __restrict__ keys) {
  float thresh = (float)(*thrBin) / (float)HBINS;
  long long stride = (long long)gridDim.x*blockDim.x;
  for (long long i = (long long)blockIdx.x*blockDim.x + threadIdx.x; i < NM; i += stride) {
    float tc = tabC[i];
    if (tc > 0.f) {
      float v = tabS[i] / fmaxf(tc, 1.f);
      if (v >= thresh) {
        unsigned pos = atomicAdd(candCnt, 1u);
        if (pos < NSORT)
          keys[pos] = ((unsigned long long)__float_as_uint(v) << 32)
                    | (unsigned long long)(~(unsigned)i);   // value desc, idx asc tie-break
      }
    }
  }
}

// ---------------- kernel 5: single-block bitonic sort, descending ----------------
__global__ void bitonic_sort(unsigned long long* __restrict__ keys) {
  int tid = threadIdx.x;
  for (unsigned k = 2; k <= NSORT; k <<= 1) {
    for (unsigned j = k >> 1; j > 0; j >>= 1) {
      for (unsigned i = tid; i < NSORT; i += blockDim.x) {
        unsigned ixj = i ^ j;
        if (ixj > i) {
          unsigned long long a = keys[i], b = keys[ixj];
          bool up = ((i & k) == 0);
          if (up ? (a < b) : (a > b)) { keys[i] = b; keys[ixj] = a; }
        }
      }
      __syncthreads();
    }
  }
}

// ---------------- kernel 6: local weighted Procrustes per block + WMMA features Y ----------------
__global__ void local_procrustes(const float* __restrict__ src_pts, const float* __restrict__ tgt_pts,
                                 const int* __restrict__ sidx, const int* __restrict__ tidx,
                                 const float* __restrict__ score, const unsigned char* __restrict__ corr,
                                 float* __restrict__ Rl, float* __restrict__ tl,
                                 float* __restrict__ cntB, _Float16* __restrict__ featY) {
  __shared__ float sp[KK][3], tp[KK][3];
  __shared__ float sums[17];
  int b = blockIdx.x, tid = threadIdx.x;
  if (tid < 17) sums[tid] = 0.f;
  if (tid < KK) {
    int id = sidx[b*KK+tid];
    sp[tid][0]=src_pts[id*3]; sp[tid][1]=src_pts[id*3+1]; sp[tid][2]=src_pts[id*3+2];
  } else if (tid < 2*KK) {
    int id = tidx[b*KK + tid - KK];
    tp[tid-KK][0]=tgt_pts[id*3]; tp[tid-KK][1]=tgt_pts[id*3+1]; tp[tid-KK][2]=tgt_pts[id*3+2];
  }
  __syncthreads();
  float acc[17];
  for (int a=0;a<17;a++) acc[a]=0.f;
  for (int r=0;r<16;r++) {
    int p = tid + 256*r;                 // 0..4095
    int i = p >> 6, j = p & 63;
    unsigned char c = corr[b*4096 + p];
    float w = c ? score[b*4096 + p] : 0.f;
    acc[16] += c ? 1.f : 0.f;
    float sx=sp[i][0], sy=sp[i][1], sz=sp[i][2];
    float tx=tp[j][0], ty=tp[j][1], tz=tp[j][2];
    acc[0] += w;
    acc[1]+=w*sx; acc[2]+=w*sy; acc[3]+=w*sz;
    acc[4]+=w*tx; acc[5]+=w*ty; acc[6]+=w*tz;
    acc[7]+=w*sx*tx;  acc[8]+=w*sx*ty;  acc[9]+=w*sx*tz;
    acc[10]+=w*sy*tx; acc[11]+=w*sy*ty; acc[12]+=w*sy*tz;
    acc[13]+=w*sz*tx; acc[14]+=w*sz*ty; acc[15]+=w*sz*tz;
  }
  int lane = tid & 31;
  for (int a=0;a<17;a++) {
    float v = warp_sum(acc[a]);
    if (lane == 0) atomicAdd(&sums[a], v);
  }
  __syncthreads();
  if (tid == 0) {
    float R[3][3], t[3];
    procrustes_from_sums(sums, R, t);
    for (int i=0;i<3;i++) for (int j=0;j<3;j++) Rl[b*9+3*i+j] = R[i][j];
    for (int i=0;i<3;i++) tl[b*3+i] = t[i];
    cntB[b] = sums[16];
    // y features: resid2(b,g) = base_g + <y_b, x_g>
    _Float16* y = featY + b*16;
    for (int i=0;i<3;i++) for (int j=0;j<3;j++) y[3*i+j] = (_Float16)(-2.f*R[i][j]);
    for (int i=0;i<3;i++) y[9+i]  = (_Float16)(-2.f*t[i]);
    for (int j=0;j<3;j++) {
      float rt = 0.f; for (int a=0;a<3;a++) rt += R[a][j]*t[a];
      y[12+j] = (_Float16)(2.f*rt);
    }
    y[15] = (_Float16)(t[0]*t[0]+t[1]*t[1]+t[2]*t[2]);
  }
}

// ---------------- kernel 7: gather top-GG, write outputs + WMMA features X ----------------
__global__ void gather_top(const unsigned long long* __restrict__ keys,
                           const float* __restrict__ src_pts, const float* __restrict__ tgt_pts,
                           float* __restrict__ dout, _Float16* __restrict__ featX,
                           float* __restrict__ baseV) {
  int i = blockIdx.x*blockDim.x + threadIdx.x;
  if (i >= GG) return;
  unsigned long long key = keys[i];
  float s[3] = {0,0,0}, t[3] = {0,0,0}; float v = 0.f;
  if (key != 0ull) {
    v = __uint_as_float((unsigned)(key >> 32));
    unsigned idx = ~(unsigned)(key & 0xffffffffu);
    unsigned si = idx / MPTS, ti = idx % MPTS;
    s[0]=src_pts[si*3]; s[1]=src_pts[si*3+1]; s[2]=src_pts[si*3+2];
    t[0]=tgt_pts[ti*3]; t[1]=tgt_pts[ti*3+1]; t[2]=tgt_pts[ti*3+2];
  }
  for (int c=0;c<3;c++) { dout[i*3+c]=s[c]; dout[GG*3+i*3+c]=t[c]; }
  dout[GG*6 + i] = v;
  _Float16* x = featX + i*16;
  for (int a=0;a<3;a++) for (int bq=0;bq<3;bq++) x[3*a+bq] = (_Float16)(t[a]*s[bq]);
  for (int a=0;a<3;a++) { x[9+a] = (_Float16)t[a]; x[12+a] = (_Float16)s[a]; }
  x[15] = (_Float16)1.f;
  baseV[i] = t[0]*t[0]+t[1]*t[1]+t[2]*t[2] + s[0]*s[0]+s[1]*s[1]+s[2]*s[2];
}

// ---------------- kernel 8: 256x4096 K=16 inlier GEMM via v_wmma_f32_16x16x32_f16 ----------------
// A (16x32 f16): rows = local-block features y_b (K padded 16->32 with 0)
// B (32x16 f16): cols = candidate features x_g
// Lane layouts: row/col = lane%16; contiguous K = 16*(lane/16) + e in both A and B.
__global__ void __launch_bounds__(128) inlier_count(const _Float16* __restrict__ featY,
                                                    const _Float16* __restrict__ featX,
                                                    const float* __restrict__ baseV,
                                                    int* __restrict__ inl) {
  int flat = blockIdx.x*128 + threadIdx.x;
  int wave = flat >> 5;
  int lane = flat & 31;
  int tb = wave >> 8;        // 0..15  (b tile)
  int tg = wave & 255;       // 0..255 (g tile)
  int g0 = lane >> 4;        // half-wave id
  int col = lane & 15;
  int brow = tb*16 + col;    // A-matrix row (M = lane%16)
  int gcol = tg*16 + col;    // B-matrix col (N = lane%16)
  v16h a, bm;
  for (int e=0;e<16;e++) {
    // A: K = e + 16*g0
    int kA = e + 16*g0;
    a[e]  = (kA < 16) ? featY[brow*16 + kA] : (_Float16)0.f;
    // B: K = e + 16*g0
    int kB = e + 16*g0;
    bm[e] = (kB < 16) ? featX[gcol*16 + kB] : (_Float16)0.f;
  }
  v8f c = {0.f,0.f,0.f,0.f,0.f,0.f,0.f,0.f};
  c = __builtin_amdgcn_wmma_f32_16x16x32_f16(false, a, false, bm, (short)0, c, false, false);
  float bg = baseV[gcol];
  for (int r=0;r<8;r++) {
    bool in = (bg + c[r]) < RAD2;         // element (M=r+8*g0, N=col)
    unsigned mm = (unsigned)__ballot(in); // wave32 mask
    if (lane == 0) {
      atomicAdd(&inl[tb*16 + r],     (int)__popc(mm & 0xFFFFu)); // lanes 0-15: M=r
      atomicAdd(&inl[tb*16 + r + 8], (int)__popc(mm >> 16));     // lanes 16-31: M=r+8
    }
  }
}

// ---------------- kernel 9: pick best local block (first max), seed transform ----------------
__global__ void pick_best(const int* __restrict__ inl, const float* __restrict__ cntB,
                          const float* __restrict__ Rl, const float* __restrict__ tl,
                          float* __restrict__ cur) {
  __shared__ int vals[256]; __shared__ int idxs[256];
  int tid = threadIdx.x;
  vals[tid] = (cntB[tid] >= 3.0f) ? inl[tid] : -1;   // MIN_LOCAL
  idxs[tid] = tid;
  __syncthreads();
  for (int s=128; s>0; s>>=1) {
    if (tid < s) {
      if (vals[tid+s] > vals[tid] || (vals[tid+s] == vals[tid] && idxs[tid+s] < idxs[tid])) {
        vals[tid]=vals[tid+s]; idxs[tid]=idxs[tid+s];
      }
    }
    __syncthreads();
  }
  if (tid == 0) {
    int b = idxs[0];
    for (int i=0;i<9;i++) cur[i]   = Rl[b*9+i];
    for (int i=0;i<3;i++) cur[9+i] = tl[b*3+i];
  }
}

// ---------------- kernel 10: one global Procrustes refinement step ----------------
__global__ void global_step(const float* __restrict__ dout, float* __restrict__ cur,
                            int write_out, float* __restrict__ Tout) {
  __shared__ float sums[16];
  int tid = threadIdx.x;
  if (tid < 16) sums[tid] = 0.f;
  __syncthreads();
  float R[3][3], tr[3];
  for (int i=0;i<3;i++) for (int j=0;j<3;j++) R[i][j] = cur[3*i+j];
  for (int i=0;i<3;i++) tr[i] = cur[9+i];
  float acc[16];
  for (int a=0;a<16;a++) acc[a]=0.f;
  for (int g = tid; g < GG; g += 256) {
    float sx=dout[g*3], sy=dout[g*3+1], sz=dout[g*3+2];
    float tx=dout[GG*3+g*3], ty=dout[GG*3+g*3+1], tz=dout[GG*3+g*3+2];
    float sc = dout[GG*6 + g];
    float ax = R[0][0]*sx + R[0][1]*sy + R[0][2]*sz + tr[0];
    float ay = R[1][0]*sx + R[1][1]*sy + R[1][2]*sz + tr[1];
    float az = R[2][0]*sx + R[2][1]*sy + R[2][2]*sz + tr[2];
    float dx=tx-ax, dy=ty-ay, dz=tz-az;
    float w = (dx*dx+dy*dy+dz*dz < RAD2) ? sc : 0.f;
    acc[0]+=w;
    acc[1]+=w*sx; acc[2]+=w*sy; acc[3]+=w*sz;
    acc[4]+=w*tx; acc[5]+=w*ty; acc[6]+=w*tz;
    acc[7]+=w*sx*tx;  acc[8]+=w*sx*ty;  acc[9]+=w*sx*tz;
    acc[10]+=w*sy*tx; acc[11]+=w*sy*ty; acc[12]+=w*sy*tz;
    acc[13]+=w*sz*tx; acc[14]+=w*sz*ty; acc[15]+=w*sz*tz;
  }
  int lane = tid & 31;
  for (int a=0;a<16;a++) {
    float v = warp_sum(acc[a]);
    if (lane == 0) atomicAdd(&sums[a], v);
  }
  __syncthreads();
  if (tid == 0) {
    float Rn[3][3], tn[3];
    procrustes_from_sums(sums, Rn, tn);
    for (int i=0;i<3;i++) for (int j=0;j<3;j++) cur[3*i+j] = Rn[i][j];
    for (int i=0;i<3;i++) cur[9+i] = tn[i];
    if (write_out) {
      for (int i=0;i<3;i++) {
        for (int j=0;j<3;j++) Tout[4*i+j] = Rn[i][j];
        Tout[4*i+3] = tn[i];
      }
      Tout[12]=0.f; Tout[13]=0.f; Tout[14]=0.f; Tout[15]=1.f;
    }
  }
}

// ---------------- host driver ----------------
extern "C" void kernel_launch(void* const* d_in, const int* in_sizes, int n_in,
                              void* d_out, int out_size, void* d_ws, size_t ws_size,
                              hipStream_t stream) {
  (void)in_sizes; (void)n_in; (void)out_size; (void)ws_size;
  const float* src_pts = (const float*)d_in[0];
  const float* tgt_pts = (const float*)d_in[1];
  const int*   sidx    = (const int*)d_in[2];
  const int*   tidx    = (const int*)d_in[3];
  const float* score   = (const float*)d_in[4];
  const unsigned char* corr = (const unsigned char*)d_in[5]; // numpy bool layout
  float* dout = (float*)d_out;

  char* w = (char*)d_ws;
  size_t off = 0;
  auto alloc = [&](size_t bytes, size_t align)->char* {
    off = (off + align - 1) & ~(align - 1);
    char* p = w + off; off += bytes; return p;
  };
  float* tabS = (float*)alloc((size_t)(NM+1)*4, 256);
  float* tabC = (float*)alloc((size_t)(NM+1)*4, 256);
  size_t metaBytes = (size_t)HBINS*4 + 4 + 4 + 256*4 + (size_t)NSORT*8;
  char* metaP = alloc(metaBytes, 256);
  unsigned* hist    = (unsigned*)metaP;
  unsigned* thrBin  = (unsigned*)(metaP + HBINS*4);
  unsigned* candCnt = thrBin + 1;
  int*      inl     = (int*)(candCnt + 1);
  unsigned long long* keys = (unsigned long long*)(metaP + HBINS*4 + 8 + 256*4);
  float* Rl    = (float*)alloc(256*9*4, 256);
  float* tl    = (float*)alloc(256*3*4, 16);
  float* cntB  = (float*)alloc(256*4, 16);
  _Float16* featY = (_Float16*)alloc(256*16*2, 16);
  _Float16* featX = (_Float16*)alloc((size_t)GG*16*2, 16);
  float* baseV = (float*)alloc((size_t)GG*4, 16);
  float* cur   = (float*)alloc(16*4, 16);

  // zero tables (span includes alignment gap) + meta (hist/counters/inl/keys)
  hipMemsetAsync(tabS, 0, (size_t)((char*)(tabC + (NM+1)) - (char*)tabS), stream);
  hipMemsetAsync(metaP, 0, metaBytes, stream);

  scatter_pairs<<<(BB*KK*KK)/256, 256, 0, stream>>>(sidx, tidx, score, corr, tabS, tabC);
  hist_kernel<<<2048, 256, 0, stream>>>(tabS, tabC, hist);
  find_threshold<<<1, 1, 0, stream>>>(hist, thrBin);
  compact_kernel<<<2048, 256, 0, stream>>>(tabS, tabC, thrBin, candCnt, keys);
  bitonic_sort<<<1, 1024, 0, stream>>>(keys);
  local_procrustes<<<BB, 256, 0, stream>>>(src_pts, tgt_pts, sidx, tidx, score, corr,
                                           Rl, tl, cntB, featY);
  gather_top<<<GG/256, 256, 0, stream>>>(keys, src_pts, tgt_pts, dout, featX, baseV);
  inlier_count<<<1024, 128, 0, stream>>>(featY, featX, baseV, inl);
  pick_best<<<1, 256, 0, stream>>>(inl, cntB, Rl, tl, cur);
  for (int s = 0; s < 5; s++)   // NUM_STEPS
    global_step<<<1, 256, 0, stream>>>(dout, cur, (s == 4) ? 1 : 0, dout + GG*7);
}
